// GAT1Custom_79748952752834
// MI455X (gfx1250) — matrix operand order repaired and checked by reading, not compiled, not run on hardware
//
#include <hip/hip_runtime.h>

// ---------------------------------------------------------------------------
// GAT layer on gfx1250 (MI455X): bf16 WMMA for all GEMM phases, VALU for the
// pairwise leaky-relu attention scores (not matmul-able), wave32 reductions.
// ---------------------------------------------------------------------------

#define N_TOTAL 2048
#define DIM     128       // N_IN == N_OUT
#define NB      8         // graphs
#define NPG     256       // nodes per graph
#define ALPHA   0.2f

typedef __attribute__((ext_vector_type(16))) __bf16       bf16x16;
typedef __attribute__((ext_vector_type(8)))  float        f32x8;
typedef __attribute__((ext_vector_type(4)))  unsigned int u32x4;

union BfFrag { bf16x16 v; u32x4 u[2]; };

// D = A(16x32) * B(32x16) + C, bf16 in / f32 out  (V_WMMA_F32_16X16X32_BF16)
__device__ __forceinline__ f32x8 wmma_bf16(bf16x16 a, bf16x16 b, f32x8 c) {
    return __builtin_amdgcn_wmma_f32_16x16x32_bf16(
        /*neg_a=*/false, a, /*neg_b=*/false, b,
        /*c_mod=*/(short)0, c, /*reuse_a=*/false, /*reuse_b=*/false);
}

// A fragment, 16-bit A 16x32 layout (ISA 7.12.2):
//   lane<16 : row = m0+lane,    K = {k0..k0+7} U {k0+16..k0+23}
//   lane>=16: row = m0+lane-16, K = {k0+8..k0+15} U {k0+24..k0+31}
__device__ __forceinline__ bf16x16
load_a_frag(const __bf16* base, int ldk, int m0, int k0, int lane) {
    const __bf16* p = base + (m0 + (lane & 15)) * ldk + k0 + ((lane & 16) ? 8 : 0);
    BfFrag f;
    f.u[0] = *(const u32x4*)(p);        // 8 bf16: K block 1
    f.u[1] = *(const u32x4*)(p + 16);   // 8 bf16: K block 2 (+16)
    return f.v;
}

// B fragment from column-major staging (bc[n][k], row stride ldk):
//   lane<16 : col = n0+lane,    K = {k0..k0+7} U {k0+16..k0+23}
//   lane>=16: col = n0+lane-16, K = {k0+8..k0+15} U {k0+24..k0+31}
__device__ __forceinline__ bf16x16
load_b_frag(const __bf16* bc, int ldk, int n0, int k0, int lane) {
    const __bf16* p = bc + (n0 + (lane & 15)) * ldk + k0 + ((lane & 16) ? 8 : 0);
    BfFrag f;
    f.u[0] = *(const u32x4*)(p);
    f.u[1] = *(const u32x4*)(p + 16);
    return f.v;
}

// C/D 16x16 f32 layout: element r of lane L -> row r + (L>=16 ? 8:0), col L&15.
__device__ __forceinline__ void
store_tile_f32(float* dst, int ld, int row0, const f32x8* acc, int lane) {
    const int rbase = row0 + ((lane & 16) ? 8 : 0);
    const int col   = lane & 15;
#pragma unroll
    for (int t = 0; t < 8; ++t)
#pragma unroll
        for (int r = 0; r < 8; ++r)
            dst[(rbase + r) * ld + t * 16 + col] = acc[t][r];
}

// ---------------------------------------------------------------------------
// Kernel 1: fused  h = x@W,  Ah = h@L1^T,  Ax = x@L2^T
// grid 16 x 256 thr (8 waves); each block owns 128 rows, each wave 16 rows.
// LDS (dynamic, 160 KB): xs, wT, l1, l2, hs  each [128][128] bf16.
// ---------------------------------------------------------------------------
__global__ __launch_bounds__(256)
void gat_linear_kernel(const float* __restrict__ x, const float* __restrict__ W,
                       const float* __restrict__ lin_w,
                       float* __restrict__ h_out, float* __restrict__ Ah_out,
                       float* __restrict__ Ax_out) {
    extern __shared__ __align__(16) char dynsmem[];
    __bf16* xs = (__bf16*)dynsmem;          // x tile, row-major [m][k]
    __bf16* wT = xs + DIM * DIM;            // W^T  : wT[o][c] = W[c][o]
    __bf16* l1 = wT + DIM * DIM;            // l1[k][o] = lin_w[k][o]
    __bf16* l2 = l1 + DIM * DIM;            // l2[k][c] = lin_w[k][128+c]
    __bf16* hs = l2 + DIM * DIM;            // h tile bf16, row-major

    const int tid  = threadIdx.x;
    const int row0 = blockIdx.x * 128;

    for (int idx = tid; idx < DIM * DIM; idx += 256)
        xs[idx] = (__bf16)x[row0 * DIM + idx];
    for (int idx = tid; idx < DIM * DIM; idx += 256) {
        const int o = idx >> 7, c = idx & 127;
        wT[idx] = (__bf16)W[c * DIM + o];
    }
    for (int idx = tid; idx < DIM * DIM; idx += 256) {
        const int k = idx >> 7, c = idx & 127;
        l1[idx] = (__bf16)lin_w[k * 2 * DIM + c];
        l2[idx] = (__bf16)lin_w[k * 2 * DIM + DIM + c];
    }
    __syncthreads();

    const int wave = tid >> 5;
    const int lane = tid & 31;
    const int m0   = wave * 16;            // local row base of this wave

    // ---- h = x @ W ----
    {
        f32x8 acc[8] = {};
#pragma unroll
        for (int k0 = 0; k0 < DIM; k0 += 32) {
            const bf16x16 a = load_a_frag(xs, DIM, m0, k0, lane);
#pragma unroll
            for (int t = 0; t < 8; ++t)
                acc[t] = wmma_bf16(a, load_b_frag(wT, DIM, t * 16, k0, lane), acc[t]);
        }
        store_tile_f32(h_out, DIM, row0 + m0, acc, lane);
        // stash bf16 copy for the Ah GEMM
        const int rbase = m0 + ((lane & 16) ? 8 : 0);
        const int col   = lane & 15;
#pragma unroll
        for (int t = 0; t < 8; ++t)
#pragma unroll
            for (int r = 0; r < 8; ++r)
                hs[(rbase + r) * DIM + t * 16 + col] = (__bf16)acc[t][r];
    }
    __syncthreads();

    // ---- Ah = h @ lin_w[:, :128]^T ----
    {
        f32x8 acc[8] = {};
#pragma unroll
        for (int k0 = 0; k0 < DIM; k0 += 32) {
            const bf16x16 a = load_a_frag(hs, DIM, m0, k0, lane);
#pragma unroll
            for (int t = 0; t < 8; ++t)
                acc[t] = wmma_bf16(a, load_b_frag(l1, DIM, t * 16, k0, lane), acc[t]);
        }
        store_tile_f32(Ah_out, DIM, row0 + m0, acc, lane);
    }
    // ---- Ax = x @ lin_w[:, 128:]^T ----
    {
        f32x8 acc[8] = {};
#pragma unroll
        for (int k0 = 0; k0 < DIM; k0 += 32) {
            const bf16x16 a = load_a_frag(xs, DIM, m0, k0, lane);
#pragma unroll
            for (int t = 0; t < 8; ++t)
                acc[t] = wmma_bf16(a, load_b_frag(l2, DIM, t * 16, k0, lane), acc[t]);
        }
        store_tile_f32(Ax_out, DIM, row0 + m0, acc, lane);
    }
}

// ---------------------------------------------------------------------------
// Kernel 2: att[i,j] = softmax_j( sum_k lrelu(Ah[i,k]+Ax[j,k]+b[k]) * a[k] )
// One wave per row i; block = 8 waves = 8 rows of one graph; grid 256.
// Axs staged in LDS with stride 129 floats -> conflict-free j-strided reads.
// ---------------------------------------------------------------------------
#define AXS_LD 129
__global__ __launch_bounds__(256)
void gat_attn_softmax_kernel(const float* __restrict__ Ah,
                             const float* __restrict__ Ax,
                             const float* __restrict__ lin_b,
                             const float* __restrict__ avec,
                             float* __restrict__ att) {
    extern __shared__ __align__(16) char dynsmem[];
    float* Axs = (float*)dynsmem;              // [256][129]  (lin_b folded in)
    float* Ahs = Axs + NPG * AXS_LD;           // [8][128]
    float* as  = Ahs + 8 * DIM;                // [128]

    const int tid   = threadIdx.x;
    const int row0  = blockIdx.x * 8;          // first global row of this block
    const int jbase = (row0 >> 8) << 8;        // graph start (n = 256)

    for (int idx = tid; idx < NPG * DIM; idx += 256) {
        const int j = idx >> 7, k = idx & 127;
        Axs[j * AXS_LD + k] = Ax[(jbase + j) * DIM + k] + lin_b[k];
    }
    for (int idx = tid; idx < 8 * DIM; idx += 256)
        Ahs[idx] = Ah[row0 * DIM + idx];
    if (tid < DIM) as[tid] = avec[tid];
    __syncthreads();

    const int wave = tid >> 5;
    const int lane = tid & 31;
    const float* ah = Ahs + wave * DIM;        // broadcast reads within wave

    float s[8];
#pragma unroll
    for (int jj = 0; jj < 8; ++jj) {
        const float* axj = Axs + (lane + jj * 32) * AXS_LD;
        float acc = 0.f;
#pragma unroll 4
        for (int k = 0; k < DIM; ++k) {
            const float z = ah[k] + axj[k];
            const float e = fmaxf(z, ALPHA * z);   // leaky_relu, 0<alpha<1
            acc = fmaf(e, as[k], acc);
        }
        s[jj] = acc;
    }
    // softmax over 256 values: 8 per lane x 32 lanes (wave32 xor reduction)
    float m = s[0];
#pragma unroll
    for (int jj = 1; jj < 8; ++jj) m = fmaxf(m, s[jj]);
#pragma unroll
    for (int off = 16; off >= 1; off >>= 1) m = fmaxf(m, __shfl_xor(m, off, 32));
    float sum = 0.f;
#pragma unroll
    for (int jj = 0; jj < 8; ++jj) { s[jj] = __expf(s[jj] - m); sum += s[jj]; }
#pragma unroll
    for (int off = 16; off >= 1; off >>= 1) sum += __shfl_xor(sum, off, 32);
    const float inv = 1.f / sum;
    const int i = row0 + wave;
#pragma unroll
    for (int jj = 0; jj < 8; ++jj)
        att[i * NPG + lane + jj * 32] = s[jj] * inv;
}

// ---------------------------------------------------------------------------
// Kernel 3: per-graph  h' = attention(256x256) @ h(256x128)  via bf16 WMMA.
// grid 16 = 8 graphs x 2 row-tiles of 128; 8 waves x 16 rows each.
// LDS 128 KB: att tile row-major + h staged transposed (K-contiguous).
// ---------------------------------------------------------------------------
__global__ __launch_bounds__(256)
void gat_aggregate_kernel(const float* __restrict__ att,
                          const float* __restrict__ h,
                          float* __restrict__ out) {
    extern __shared__ __align__(16) char dynsmem[];
    __bf16* atts = (__bf16*)dynsmem;        // [128][256]  A rows
    __bf16* hT   = atts + 128 * NPG;        // [128][256]  hT[o][j] = h[j][o]

    const int tid   = threadIdx.x;
    const int b     = blockIdx.x >> 1;
    const int row0  = b * NPG + (blockIdx.x & 1) * 128;  // output rows
    const int jbase = b * NPG;

    for (int idx = tid; idx < 128 * NPG; idx += 256)
        atts[idx] = (__bf16)att[row0 * NPG + idx];
    for (int idx = tid; idx < NPG * DIM; idx += 256) {
        const int j = idx >> 7, o = idx & 127;
        hT[o * NPG + j] = (__bf16)h[(jbase + j) * DIM + o];
    }
    __syncthreads();

    const int wave = tid >> 5;
    const int lane = tid & 31;
    const int m0   = wave * 16;

    f32x8 acc[8] = {};
#pragma unroll
    for (int k0 = 0; k0 < NPG; k0 += 32) {          // K = 256 neighbors
        const bf16x16 a = load_a_frag(atts, NPG, m0, k0, lane);
#pragma unroll
        for (int t = 0; t < 8; ++t)
            acc[t] = wmma_bf16(a, load_b_frag(hT, NPG, t * 16, k0, lane), acc[t]);
    }
    store_tile_f32(out, DIM, row0 + m0, acc, lane);
}

// ---------------------------------------------------------------------------
extern "C" void kernel_launch(void* const* d_in, const int* in_sizes, int n_in,
                              void* d_out, int out_size, void* d_ws, size_t ws_size,
                              hipStream_t stream) {
    (void)in_sizes; (void)n_in; (void)out_size; (void)ws_size;
    const float* x     = (const float*)d_in[0];
    // d_in[1] = batch (segment ids): contiguous blocks of 256, not needed
    const float* W     = (const float*)d_in[2];
    const float* lin_w = (const float*)d_in[3];
    const float* lin_b = (const float*)d_in[4];
    const float* avec  = (const float*)d_in[5];
    float* out = (float*)d_out;

    float* h   = (float*)d_ws;                 // [2048][128] f32
    float* Ah  = h  + N_TOTAL * DIM;           // [2048][128]
    float* Ax  = Ah + N_TOTAL * DIM;           // [2048][128]
    float* att = Ax + N_TOTAL * DIM;           // [2048][256]  (~5 MB total)

    const size_t smem1 = (size_t)5 * DIM * DIM * sizeof(__bf16);              // 160 KB
    const size_t smem2 = (size_t)(NPG * AXS_LD + 8 * DIM + DIM) * sizeof(float); // ~134 KB
    const size_t smem3 = (size_t)2 * 128 * NPG * sizeof(__bf16);              // 128 KB

    gat_linear_kernel<<<N_TOTAL / 128, 256, smem1, stream>>>(x, W, lin_w, h, Ah, Ax);
    gat_attn_softmax_kernel<<<N_TOTAL / 8, 256, smem2, stream>>>(Ah, Ax, lin_b, avec, att);
    gat_aggregate_kernel<<<NB * 2, 256, smem3, stream>>>(att, h, out);
}
